// EfficientFormerMeta3D_16655883174069
// MI455X (gfx1250) — matrix operand order repaired and checked by reading, not compiled, not run on hardware
//
#include <hip/hip_runtime.h>
#include <math.h>

// ---------------------------------------------------------------------------
// EfficientFormer Meta3D block on MI455X (gfx1250), bf16 WMMA pipeline.
// B=1024 S=49 DIM=448; M = B*S = 50176 (= 196*256). All GEMM dims are exact
// multiples of the 256x64x32 tiling -> no edge handling, EXEC all-ones.
// GEMM: 256x64 block tile, 8 waves, 32x64 wave tile (8 WMMA / K-step / wave,
// 12 ds_load_b128 / 8 WMMA), double-buffered LDS (one barrier per K-step).
// ---------------------------------------------------------------------------

#define MTOT        (1024 * 49)   // 50176
#define DIMC        448
#define QKVO        1536
#define TOTEXP      1024
#define MLPH        1792
#define NHEADS      8
#define SEQ         49
#define SPAD        64
#define HSTRIDE     192           // per-head qkv chunk: 32 q + 32 k + 128 v
#define EXPD        128
#define NOFF        49
#define ATT_SCALE   0.17677669529663687f   // 32^-0.5

typedef __attribute__((ext_vector_type(16))) __bf16 bf16x16;
typedef __attribute__((ext_vector_type(8)))  float  f32x8;

union FragBF { bf16x16 v; uint4 q[2]; };

__device__ __forceinline__ unsigned short f32_to_bf16(float f) {
    unsigned int u = __float_as_uint(f);
    unsigned int lsb = (u >> 16) & 1u;
    u += 0x7fffu + lsb;                  // round-to-nearest-even
    return (unsigned short)(u >> 16);
}

__device__ __forceinline__ f32x8 wmma_bf16(const FragBF& a, const FragBF& b, f32x8 c) {
    return __builtin_amdgcn_wmma_f32_16x16x32_bf16(
        false, a.v, false, b.v, (short)0, c, false, false);
}

// ---------------------------------------------------------------------------
// f32 -> bf16 elementwise (weight conversion)
// ---------------------------------------------------------------------------
__global__ void cvt_f32_bf16(const float* __restrict__ in,
                             unsigned short* __restrict__ out, int n) {
    int i = blockIdx.x * blockDim.x + threadIdx.x;
    if (i < n) out[i] = f32_to_bf16(in[i]);
}

// ---------------------------------------------------------------------------
// LayerNorm over DIMC=448, output bf16. One 64-thread block per row.
// ---------------------------------------------------------------------------
__global__ __launch_bounds__(64)
void ln_bf16_kernel(const float* __restrict__ x, const float* __restrict__ g,
                    const float* __restrict__ bb, unsigned short* __restrict__ y) {
    __shared__ float r0[64];
    __shared__ float r1[64];
    const int row = blockIdx.x;
    const int t = threadIdx.x;
    const float* xr = x + (size_t)row * DIMC;
    float v[7];
    float s = 0.f, ss = 0.f;
#pragma unroll
    for (int i = 0; i < 7; ++i) {
        v[i] = xr[t + 64 * i];
        s += v[i]; ss += v[i] * v[i];
    }
    r0[t] = s; r1[t] = ss;
    __syncthreads();
    for (int off = 32; off > 0; off >>= 1) {
        if (t < off) { r0[t] += r0[t + off]; r1[t] += r1[t + off]; }
        __syncthreads();
    }
    const float mean = r0[0] * (1.0f / DIMC);
    const float var  = r1[0] * (1.0f / DIMC) - mean * mean;
    const float inv  = rsqrtf(var + 1e-5f);
#pragma unroll
    for (int i = 0; i < 7; ++i) {
        int c = t + 64 * i;
        y[(size_t)row * DIMC + c] = f32_to_bf16((v[i] - mean) * inv * g[c] + bb[c]);
    }
}

// ---------------------------------------------------------------------------
// bf16 GEMM  C[M,N] = A[M,K] * B[K,N]  with templated epilogue.
//   MODE 1: out_bf16 = acc + bias                    (QKV)
//   MODE 2: out_bf16 = gelu(acc + bias)              (MLP up)
//   MODE 3: out_f32  = resid + ls * (acc + bias)     (proj / MLP down)
// Block tile 256x64, K-step 32, 256 threads = 8 waves, wave tile 32x64,
// double-buffered LDS.
// ---------------------------------------------------------------------------
template <int MODE>
__global__ __launch_bounds__(256)
void gemm_bf16(const unsigned short* __restrict__ A,
               const unsigned short* __restrict__ Bm,
               const float* __restrict__ bias,
               const float* __restrict__ ls,
               const float* __restrict__ resid,
               unsigned short* __restrict__ outB,
               float* __restrict__ outF,
               int M, int N, int K) {
    __shared__ unsigned short sA[2][256 * 32];   // [m][k] row-major
    __shared__ unsigned short sB[2][64 * 32];    // [n][k] (transposed)

    const int tid  = threadIdx.x;
    const int wave = tid >> 5;
    const int lane = tid & 31;
    const int m0 = blockIdx.x * 256;
    const int n0 = blockIdx.y * 64;

    // staging coordinates
    const int arow = tid >> 1;            // 0..127 (and +128)
    const int akh  = (tid & 1) * 16;      // half-row of 16 bf16
    const int bkr  = tid >> 3;            // 0..31
    const int bnc  = (tid & 7) * 8;       // 8 cols

    auto stage = [&](int k0, int buf) {
        // A tile 256x32: each thread stages two half-rows (2x2 uint4)
#pragma unroll
        for (int rr = 0; rr < 2; ++rr) {
            const int row = arow + rr * 128;
            const uint4* gp = (const uint4*)(A + (size_t)(m0 + row) * K + k0 + akh);
            *(uint4*)(sA[buf] + row * 32 + akh)     = gp[0];
            *(uint4*)(sA[buf] + row * 32 + akh + 8) = gp[1];
        }
        // B tile 32x64 from row-major [K][N], stored transposed [n][k]
        uint4 d = *(const uint4*)(Bm + (size_t)(k0 + bkr) * N + n0 + bnc);
        const unsigned short* ds = (const unsigned short*)&d;
#pragma unroll
        for (int i = 0; i < 8; ++i) sB[buf][(bnc + i) * 32 + bkr] = ds[i];
    };

    f32x8 acc[2][4] = {};

    stage(0, 0);
    __syncthreads();

    int cur = 0;
    for (int k0 = 0; k0 < K; k0 += 32) {
        const int nxt = cur ^ 1;
        if (k0 + 32 < K) stage(k0 + 32, nxt);   // overlap next-tile loads

        // ---- A fragments: rows wave*32 + {0,16} + (lane&15) ----
        FragBF a0, a1;
        {
            const unsigned short* pa0 =
                sA[cur] + (wave * 32 + (lane & 15)) * 32 + ((lane >> 4) * 16);
            const unsigned short* pa1 = pa0 + 16 * 32;
            a0.q[0] = *(const uint4*)pa0;
            a0.q[1] = *(const uint4*)(pa0 + 8);
            a1.q[0] = *(const uint4*)pa1;
            a1.q[1] = *(const uint4*)(pa1 + 8);
        }
        // ---- 4 B fragments, each reused by both A fragments: 8 WMMAs ----
#pragma unroll
        for (int j = 0; j < 4; ++j) {
            FragBF b;
            const unsigned short* pb =
                sB[cur] + (j * 16 + (lane & 15)) * 32 + ((lane >> 4) * 16);
            b.q[0] = *(const uint4*)pb;
            b.q[1] = *(const uint4*)(pb + 8);
            acc[0][j] = wmma_bf16(a0, b, acc[0][j]);
            acc[1][j] = wmma_bf16(a1, b, acc[1][j]);
        }
        __syncthreads();
        cur = nxt;
    }

    // ---- epilogue: C tile layout -> (row, col) scatter ----
#pragma unroll
    for (int i = 0; i < 2; ++i) {
#pragma unroll
        for (int j = 0; j < 4; ++j) {
            const int col = n0 + j * 16 + (lane & 15);
            const float bv = bias[col];
#pragma unroll
            for (int v = 0; v < 8; ++v) {
                const int row = m0 + wave * 32 + i * 16 + ((lane < 16) ? v : v + 8);
                const float val = acc[i][j][v] + bv;
                if constexpr (MODE == 1) {
                    outB[(size_t)row * N + col] = f32_to_bf16(val);
                } else if constexpr (MODE == 2) {
                    const float gl = 0.5f * val * (1.0f + erff(val * 0.70710678118654752f));
                    outB[(size_t)row * N + col] = f32_to_bf16(gl);
                } else {
                    outF[(size_t)row * N + col] =
                        resid[(size_t)row * N + col] + ls[col] * val;
                }
            }
        }
    }
}

// ---------------------------------------------------------------------------
// Attention: one block per (batch, head). 128 threads = 4 waves.
// S=49 padded to 64. scores = Q*K^T (1 WMMA K-step), softmax+bias in f32,
// ctx = P*V (2 WMMA K-steps, 8 N-tiles). ctx written as bf16 for proj GEMM.
// ---------------------------------------------------------------------------
__global__ __launch_bounds__(128)
void attn_kernel(const unsigned short* __restrict__ qkv,   // [M, 1536] bf16
                 const float* __restrict__ attn_biases,    // [8, 49]
                 const int* __restrict__ bias_idxs,        // [49, 49]
                 unsigned short* __restrict__ ctx) {       // [M, 1024] bf16
    __shared__ unsigned short sQ[SPAD * 32];    // [s][d]
    __shared__ unsigned short sK[SPAD * 32];    // [s][d]  (== B frag for Q*K^T)
    __shared__ unsigned short sVt[EXPD * SPAD]; // [d][s]  (transposed V)
    __shared__ float          sS[SPAD * SPAD];  // scores f32
    __shared__ unsigned short sP[SPAD * SPAD];  // probs bf16
    __shared__ float          sBias[64];

    const int b = blockIdx.x >> 3;
    const int h = blockIdx.x & 7;
    const int tid  = threadIdx.x;
    const int wave = tid >> 5;
    const int lane = tid & 31;
    const size_t base = (size_t)b * SEQ * QKVO + (size_t)h * HSTRIDE;

    // ---- stage Q,K (64x32 each, zero-padded rows 49..63) ----
    {
        const int row = tid >> 1;
        const int kh  = (tid & 1) * 16;
        uint4 q0, q1, k0, k1;
        q0.x=q0.y=q0.z=q0.w=0; q1=q0; k0=q0; k1=q0;
        if (row < SEQ) {
            const uint4* gq = (const uint4*)(qkv + base + (size_t)row * QKVO + kh);
            const uint4* gk = (const uint4*)(qkv + base + (size_t)row * QKVO + 32 + kh);
            q0 = gq[0]; q1 = gq[1];
            k0 = gk[0]; k1 = gk[1];
        }
        *(uint4*)(sQ + row * 32 + kh)     = q0;
        *(uint4*)(sQ + row * 32 + kh + 8) = q1;
        *(uint4*)(sK + row * 32 + kh)     = k0;
        *(uint4*)(sK + row * 32 + kh + 8) = k1;
    }
    // ---- stage V transposed: sVt[d][s] ----
    for (int g = tid; g < SPAD * 16; g += 128) {
        const int s  = g >> 4;
        const int d0 = (g & 15) * 8;
        uint4 d; d.x = d.y = d.z = d.w = 0;
        if (s < SEQ)
            d = *(const uint4*)(qkv + base + (size_t)s * QKVO + 64 + d0);
        const unsigned short* p = (const unsigned short*)&d;
#pragma unroll
        for (int i = 0; i < 8; ++i) sVt[(d0 + i) * SPAD + s] = p[i];
    }
    if (tid < NOFF) sBias[tid] = attn_biases[h * NOFF + tid];
    __syncthreads();

    // ---- scores: wave w handles rows 16w..16w+15; K=32 single WMMA step ----
    {
        FragBF a;
        const unsigned short* pa =
            sQ + (wave * 16 + (lane & 15)) * 32 + ((lane >> 4) * 16);
        a.q[0] = *(const uint4*)pa;
        a.q[1] = *(const uint4*)(pa + 8);
#pragma unroll
        for (int j = 0; j < 4; ++j) {
            FragBF bf;
            const unsigned short* pb =
                sK + (j * 16 + (lane & 15)) * 32 + ((lane >> 4) * 16);
            bf.q[0] = *(const uint4*)pb;
            bf.q[1] = *(const uint4*)(pb + 8);
            f32x8 c = {};
            c = wmma_bf16(a, bf, c);
#pragma unroll
            for (int v = 0; v < 8; ++v) {
                const int row = wave * 16 + ((lane < 16) ? v : v + 8);
                sS[row * SPAD + j * 16 + (lane & 15)] = c[v];
            }
        }
    }
    __syncthreads();

    // ---- softmax with relative-position bias (one row per thread) ----
    if (tid < SPAD) {
        const int r = tid;
        if (r < SEQ) {
            const int* idxr = bias_idxs + r * NOFF;
            float mx = -1e30f;
            for (int j = 0; j < SEQ; ++j) {
                const float s = sS[r * SPAD + j] * ATT_SCALE + sBias[idxr[j]];
                sS[r * SPAD + j] = s;
                mx = fmaxf(mx, s);
            }
            float sum = 0.f;
            for (int j = 0; j < SEQ; ++j) {
                const float e = __expf(sS[r * SPAD + j] - mx);
                sS[r * SPAD + j] = e;
                sum += e;
            }
            const float inv = 1.0f / sum;
            for (int j = 0; j < SEQ; ++j)
                sP[r * SPAD + j] = f32_to_bf16(sS[r * SPAD + j] * inv);
            for (int j = SEQ; j < SPAD; ++j) sP[r * SPAD + j] = 0;
        } else {
            for (int j = 0; j < SPAD; ++j) sP[r * SPAD + j] = 0;
        }
    }
    __syncthreads();

    // ---- ctx = P * V : rows 16w..16w+15, 8 N-tiles, K = 64 (2 steps) ----
    {
        f32x8 c[8] = {};
#pragma unroll
        for (int ks = 0; ks < 2; ++ks) {
            FragBF a;
            const unsigned short* pa =
                sP + (wave * 16 + (lane & 15)) * SPAD + ks * 32 + ((lane >> 4) * 16);
            a.q[0] = *(const uint4*)pa;
            a.q[1] = *(const uint4*)(pa + 8);
#pragma unroll
            for (int j = 0; j < 8; ++j) {
                FragBF bf;
                const unsigned short* pb =
                    sVt + (j * 16 + (lane & 15)) * SPAD + ks * 32 + ((lane >> 4) * 16);
                bf.q[0] = *(const uint4*)pb;
                bf.q[1] = *(const uint4*)(pb + 8);
                c[j] = wmma_bf16(a, bf, c[j]);
            }
        }
#pragma unroll
        for (int j = 0; j < 8; ++j) {
#pragma unroll
            for (int v = 0; v < 8; ++v) {
                const int row = wave * 16 + ((lane < 16) ? v : v + 8);
                if (row < SEQ) {
                    const size_t o = ((size_t)(b * SEQ + row)) * TOTEXP +
                                     h * EXPD + j * 16 + (lane & 15);
                    ctx[o] = f32_to_bf16(c[j][v]);
                }
            }
        }
    }
}

// ---------------------------------------------------------------------------
// Launcher
// ---------------------------------------------------------------------------
extern "C" void kernel_launch(void* const* d_in, const int* in_sizes, int n_in,
                              void* d_out, int out_size, void* d_ws, size_t ws_size,
                              hipStream_t stream) {
    const float* hidden  = (const float*)d_in[0];
    const float* qkv_w   = (const float*)d_in[1];
    const float* qkv_b   = (const float*)d_in[2];
    const float* proj_w  = (const float*)d_in[3];
    const float* proj_b  = (const float*)d_in[4];
    const float* ln1_g   = (const float*)d_in[5];
    const float* ln1_b   = (const float*)d_in[6];
    const float* ln2_g   = (const float*)d_in[7];
    const float* ln2_b   = (const float*)d_in[8];
    const float* mlp_w1  = (const float*)d_in[9];
    const float* mlp_b1  = (const float*)d_in[10];
    const float* mlp_w2  = (const float*)d_in[11];
    const float* mlp_b2  = (const float*)d_in[12];
    const float* abias   = (const float*)d_in[13];
    const float* ls1     = (const float*)d_in[14];
    const float* ls2     = (const float*)d_in[15];
    const int*   bidx    = (const int*)d_in[16];

    char* ws = (char*)d_ws;
    size_t off = 0;
    auto carve = [&](size_t bytes) -> char* {
        char* p = ws + off;
        off = (off + bytes + 255) & ~(size_t)255;
        return p;
    };
    unsigned short* wq  = (unsigned short*)carve((size_t)DIMC * QKVO * 2);
    unsigned short* wp  = (unsigned short*)carve((size_t)TOTEXP * DIMC * 2);
    unsigned short* w1  = (unsigned short*)carve((size_t)DIMC * MLPH * 2);
    unsigned short* w2  = (unsigned short*)carve((size_t)MLPH * DIMC * 2);
    unsigned short* hbf = (unsigned short*)carve((size_t)MTOT * DIMC * 2);
    unsigned short* qkv = (unsigned short*)carve((size_t)MTOT * QKVO * 2);
    unsigned short* ctx = (unsigned short*)carve((size_t)MTOT * TOTEXP * 2);
    float*          x1  = (float*)carve((size_t)MTOT * DIMC * 4);
    unsigned short* m1  = (unsigned short*)carve((size_t)MTOT * MLPH * 2);

    // 1) weights -> bf16
    {
        int n;
        n = DIMC * QKVO;   cvt_f32_bf16<<<(n + 255) / 256, 256, 0, stream>>>(qkv_w,  wq, n);
        n = TOTEXP * DIMC; cvt_f32_bf16<<<(n + 255) / 256, 256, 0, stream>>>(proj_w, wp, n);
        n = DIMC * MLPH;   cvt_f32_bf16<<<(n + 255) / 256, 256, 0, stream>>>(mlp_w1, w1, n);
        n = MLPH * DIMC;   cvt_f32_bf16<<<(n + 255) / 256, 256, 0, stream>>>(mlp_w2, w2, n);
    }
    // 2) LN1 -> bf16
    ln_bf16_kernel<<<MTOT, 64, 0, stream>>>(hidden, ln1_g, ln1_b, hbf);
    // 3) QKV GEMM (bias, bf16 out)
    gemm_bf16<1><<<dim3(MTOT / 256, QKVO / 64), 256, 0, stream>>>(
        hbf, wq, qkv_b, nullptr, nullptr, qkv, nullptr, MTOT, QKVO, DIMC);
    // 4) attention
    attn_kernel<<<1024 * NHEADS, 128, 0, stream>>>(qkv, abias, bidx, ctx);
    // 5) proj GEMM + ls1 residual -> x1 (f32)
    gemm_bf16<3><<<dim3(MTOT / 256, DIMC / 64), 256, 0, stream>>>(
        ctx, wp, proj_b, ls1, hidden, nullptr, x1, MTOT, DIMC, TOTEXP);
    // 6) LN2 -> bf16
    ln_bf16_kernel<<<MTOT, 64, 0, stream>>>(x1, ln2_g, ln2_b, hbf);
    // 7) MLP up GEMM + GELU (bf16 out)
    gemm_bf16<2><<<dim3(MTOT / 256, MLPH / 64), 256, 0, stream>>>(
        hbf, w1, mlp_b1, nullptr, nullptr, m1, nullptr, MTOT, MLPH, DIMC);
    // 8) MLP down GEMM + ls2 residual -> d_out (f32)
    gemm_bf16<3><<<dim3(MTOT / 256, DIMC / 64), 256, 0, stream>>>(
        m1, w2, mlp_b2, ls2, x1, nullptr, (float*)d_out, MTOT, DIMC, MLPH);

    (void)in_sizes; (void)n_in; (void)out_size; (void)ws_size;
}
